// DeformConv2D_2551210574029
// MI455X (gfx1250) — matrix operand order repaired and checked by reading, not compiled, not run on hardware
//
#include <hip/hip_runtime.h>

// ---------------------------------------------------------------------------
// Deformable conv forward for MI455X (gfx1250, wave32, WMMA).
//   Stage A: pack off_w / conv_w into bf16 WMMA A-fragment tiles (tiny kernel)
//   Stage B: offset conv as fused im2col(LDS) + v_wmma_f32_16x16x32_bf16
//   Stage C: bilinear sampling fused with the stride-3 output conv, which is a
//            pure GEMM (stride == kernel => each sample used exactly once):
//            M=128(oc) x K=576(tap*64+ic) x N=131072(b,oi,oj)
// ---------------------------------------------------------------------------

typedef __attribute__((ext_vector_type(16))) __bf16 v16bf;
typedef __attribute__((ext_vector_type(8)))  float  v8f;

#define HH   128
#define WW   128
#define CIN  64
#define BATCH 8
#define KDIM 576                    // 9 taps * 64 channels
#define NPIX (BATCH*HH*WW)          // 131072
#define PPB  64                     // pixels per block (4 n-tiles of 16)
#define LDSB (PPB*KDIM*2)           // 73728 bytes of bf16 B-fragments

__device__ __forceinline__ unsigned short f2bf(float f) {
  unsigned int u = __float_as_uint(f);
  return (unsigned short)((u + 0x7FFFu + ((u >> 16) & 1u)) >> 16);  // RNE
}

// Pack two floats into two bf16 (lo = a, hi = b) with round-to-nearest
// (ties away). 2x v_add + 1x v_perm_b32 = 3 VALU ops per pair, vs ~7 for
// the exact-RNE bit sequence. Tie cases differ from RNE by one bf16 ULP.
#if __has_builtin(__builtin_amdgcn_perm)
__device__ __forceinline__ unsigned int pack_bf2(float a, float b) {
  unsigned int ua = __float_as_uint(a) + 0x8000u;
  unsigned int ub = __float_as_uint(b) + 0x8000u;
  // bytes 0..3 from src1 (ua), 4..7 from src0 (ub): take ua[3:2], ub[3:2]
  return __builtin_amdgcn_perm(ub, ua, 0x07060302u);
}
#else
__device__ __forceinline__ unsigned int pack_bf2(float a, float b) {
  return (unsigned)f2bf(a) | ((unsigned)f2bf(b) << 16);
}
#endif

__device__ __forceinline__ v8f zero8() {
  v8f z;
#pragma unroll
  for (int r = 0; r < 8; ++r) z[r] = 0.0f;
  return z;
}

// Byte offset of A[ row r (0..15), kin (0..31) ] inside a 1KB 16x32 A tile,
// contiguous K order (lane = r + 16*(kin>>4), element = kin & 15).
__device__ __forceinline__ int a_elem_off(int r, int kin) {
  int hA   = kin >> 4;           // lane half selector
  int lane = r + (hA << 4);
  return lane * 32 + (kin & 15) * 2;
}

// ---------------------------------------------------------------------------
// Stage A: pack weights to bf16 WMMA-A tiles. K order: k = tap*64 + ic.
// conv_w -> 8 m-tiles x 18 k-tiles; off_w -> 2 m-tiles (oc padded 18->32) x 18.
// ---------------------------------------------------------------------------
__global__ __launch_bounds__(256) void pack_weights(
    const float* __restrict__ off_w, const float* __restrict__ conv_w,
    unsigned short* __restrict__ pA_conv, unsigned short* __restrict__ pA_off) {
  int idx = blockIdx.x * 256 + threadIdx.x;
  const int n_conv = 128 * KDIM;
  const int n_off  = 32 * KDIM;
  if (idx < n_conv) {
    int k = idx % KDIM, oc = idx / KDIM;
    int m = k / 64, ic = k % 64;
    float v = conv_w[(oc * 64 + ic) * 9 + m];
    int mt = oc >> 4, r = oc & 15, kt = k >> 5, kin = k & 31;
    pA_conv[((mt * 18 + kt) * 1024 + a_elem_off(r, kin)) >> 1] = f2bf(v);
  } else if (idx < n_conv + n_off) {
    int e = idx - n_conv;
    int k = e % KDIM, oc = e / KDIM;              // oc 0..31 (18..31 zero pad)
    int m = k / 64, ic = k % 64;
    float v = (oc < 18) ? off_w[(oc * 64 + ic) * 9 + m] : 0.0f;
    int mt = oc >> 4, r = oc & 15, kt = k >> 5, kin = k & 31;
    pA_off[((mt * 18 + kt) * 1024 + a_elem_off(r, kin)) >> 1] = f2bf(v);
  }
}

// Write a 16-channel bf16 run into the LDS B-fragment tile layout.
// B 32x16: lane = 16*(kin>=16) + n%16; within lane, K pairs packed per dword.
__device__ __forceinline__ void lds_write_group(unsigned char* smem, int ntl,
                                                int kt, int laneField,
                                                const unsigned int* w32) {
  uint4* dst = (uint4*)(smem + (((ntl * 18 + kt) << 10) + laneField * 32));
  dst[0] = make_uint4(w32[0], w32[1], w32[2], w32[3]);
  dst[1] = make_uint4(w32[4], w32[5], w32[6], w32[7]);
}

// ---------------------------------------------------------------------------
// Stage B: offset conv (64->18 ch, 3x3, pad 1) as im2col-in-LDS + WMMA.
// Each block: 64 pixels. 8 waves: wave w -> (mt = w>>2, ntile = w&3).
// ---------------------------------------------------------------------------
__global__ __launch_bounds__(256) void offset_conv_wmma(
    const float* __restrict__ x, const unsigned short* __restrict__ pA_off,
    const float* __restrict__ off_b, float* __restrict__ offs) {
  __shared__ __align__(32) unsigned char smem[LDSB];
  int tid = threadIdx.x;
  int pix0 = blockIdx.x * PPB;

  for (int task = tid; task < PPB * 9; task += 256) {
    int p = task / 9, m = task % 9;
    int n = pix0 + p;
    int b = n >> 14, rem = n & 16383, i = rem >> 7, j = rem & 127;
    int ii = i + m / 3 - 1, jj = j + m % 3 - 1;
    bool ok = ((unsigned)ii < 128u) && ((unsigned)jj < 128u);
    const float* xp = x + (size_t)b * 64 * 16384 + ii * 128 + jj;
    int ntl = p >> 4, nin = p & 15;
#pragma unroll
    for (int g = 0; g < 4; ++g) {                 // 4 groups of 16 channels
      unsigned int w32[8];
#pragma unroll
      for (int c2 = 0; c2 < 8; ++c2) {
        int ic0 = g * 16 + c2 * 2;
        float v0 = ok ? xp[(size_t)ic0 * 16384] : 0.0f;
        float v1 = ok ? xp[(size_t)(ic0 + 1) * 16384] : 0.0f;
        w32[c2] = pack_bf2(v0, v1);
      }
      lds_write_group(smem, ntl, m * 2 + (g >> 1), ((g & 1) << 4) + nin, w32);
    }
  }
  __syncthreads();

  int w = tid >> 5, lane = tid & 31;
  int mt = w >> 2, ntl = w & 3;
  v8f c = zero8();
  const unsigned short* aBase = pA_off + (size_t)mt * 18 * 512 + lane * 16;
  for (int kt = 0; kt < 18; ++kt) {
    v16bf a  = *(const v16bf*)(aBase + kt * 512);
    v16bf bf = *(const v16bf*)(smem + (((ntl * 18 + kt) << 10) + lane * 32));
    c = __builtin_amdgcn_wmma_f32_16x16x32_bf16(false, a, false, bf,
                                                (short)0, c, false, false);
  }
  int h = lane >> 4, nin = lane & 15;
  int n = pix0 + ntl * 16 + nin;
  int b = n >> 14, rem = n & 16383;
  int ocb = mt * 16 + h * 8;
#pragma unroll
  for (int r = 0; r < 8; ++r) {
    int oc = ocb + r;
    if (oc < 18) offs[((size_t)b * 18 + oc) * 16384 + rem] = c[r] + off_b[oc];
  }
}

// ---------------------------------------------------------------------------
// Stage C: bilinear sampling fused with the stride-3 output conv (pure GEMM).
// Output tap (di,dj) at (oi,oj) maps to sample cell/kernel:
//   d==0 -> cell-1, k=2 (zero at boundary); d==1 -> cell, k=0; d==2 -> cell, k=1
// ---------------------------------------------------------------------------
__global__ __launch_bounds__(256) void deform_gemm_wmma(
    const float* __restrict__ x, const float* __restrict__ offs,
    const unsigned short* __restrict__ pA_conv,
    const float* __restrict__ conv_b, float* __restrict__ out) {
  __shared__ __align__(32) unsigned char smem[LDSB];
  int tid = threadIdx.x;
  int pix0 = blockIdx.x * PPB;

  for (int task = tid; task < PPB * 9; task += 256) {
    int p = task / 9, m = task % 9;
    int n = pix0 + p;
    int b = n >> 14, rem = n & 16383, oi = rem >> 7, oj = rem & 127;
    int di = m / 3, dj = m % 3;
    int ci, ki, cj, kj;
    bool valid = true;
    if (di == 0) { ci = oi - 1; ki = 2; if (ci < 0) valid = false; }
    else         { ci = oi;     ki = di - 1; }
    if (dj == 0) { cj = oj - 1; kj = 2; if (cj < 0) valid = false; }
    else         { cj = oj;     kj = dj - 1; }

    float glt = 0.f, grb = 0.f, glb = 0.f, grt = 0.f;
    int i00 = 0, i01 = 0, i10 = 0, i11 = 0;
    if (valid) {
      int ns = ki * 3 + kj;
      size_t ob = (size_t)b * 18 * 16384 + (size_t)ci * 128 + cj;
      float px = (float)ci + offs[ob + (size_t)ns * 16384];
      float py = (float)cj + offs[ob + (size_t)(9 + ns) * 16384];
      int qx = (int)floorf(px), qy = (int)floorf(py);
      int qltx = min(max(qx, 0), 127),     qlty = min(max(qy, 0), 127);
      int qrbx = min(max(qx + 1, 0), 127), qrby = min(max(qy + 1, 0), 127);
      float pxc = fminf(fmaxf(px, 0.f), 127.f);
      float pyc = fminf(fmaxf(py, 0.f), 127.f);
      float wlx = 1.f + ((float)qltx - pxc), wrx = 1.f - ((float)qrbx - pxc);
      float wly = 1.f + ((float)qlty - pyc), wry = 1.f - ((float)qrby - pyc);
      glt = wlx * wly; grb = wrx * wry; glb = wlx * wry; grt = wrx * wly;
      i00 = qltx * 128 + qlty; i11 = qrbx * 128 + qrby;
      i01 = qltx * 128 + qrby; i10 = qrbx * 128 + qlty;
    }
    const float* xb = x + (size_t)b * 64 * 16384;
    int ntl = p >> 4, nin = p & 15;
#pragma unroll
    for (int g = 0; g < 4; ++g) {
      unsigned int w32[8];
#pragma unroll
      for (int c2 = 0; c2 < 8; ++c2) {
        int ic0 = g * 16 + c2 * 2;
        const float* x0 = xb + (size_t)ic0 * 16384;
        const float* x1 = x0 + 16384;
        float v0 = fmaf(glt, x0[i00], fmaf(grb, x0[i11],
                   fmaf(glb, x0[i01], grt * x0[i10])));
        float v1 = fmaf(glt, x1[i00], fmaf(grb, x1[i11],
                   fmaf(glb, x1[i01], grt * x1[i10])));
        w32[c2] = pack_bf2(v0, v1);
      }
      lds_write_group(smem, ntl, m * 2 + (g >> 1), ((g & 1) << 4) + nin, w32);
    }
  }
  __syncthreads();

  int w = tid >> 5, lane = tid & 31;
  int mt = w;                                      // wave -> m-tile (16 oc)
  v8f c0 = zero8(), c1 = zero8(), c2 = zero8(), c3 = zero8();
  const unsigned short* aBase = pA_conv + (size_t)mt * 18 * 512 + lane * 16;
  for (int kt = 0; kt < 18; ++kt) {
    v16bf a = *(const v16bf*)(aBase + kt * 512);   // A reused for 4 n-tiles
    const unsigned char* bb = smem + (kt << 10) + lane * 32;
    v16bf b0 = *(const v16bf*)(bb);
    v16bf b1 = *(const v16bf*)(bb + 18 * 1024);
    v16bf b2 = *(const v16bf*)(bb + 36 * 1024);
    v16bf b3 = *(const v16bf*)(bb + 54 * 1024);
    c0 = __builtin_amdgcn_wmma_f32_16x16x32_bf16(false, a, false, b0, (short)0, c0, false, false);
    c1 = __builtin_amdgcn_wmma_f32_16x16x32_bf16(false, a, false, b1, (short)0, c1, false, false);
    c2 = __builtin_amdgcn_wmma_f32_16x16x32_bf16(false, a, false, b2, (short)0, c2, false, false);
    c3 = __builtin_amdgcn_wmma_f32_16x16x32_bf16(false, a, false, b3, (short)0, c3, false, false);
  }
  int h = lane >> 4, nin = lane & 15;
  int ocb = mt * 16 + h * 8;
#pragma unroll
  for (int ntl = 0; ntl < 4; ++ntl) {
    int n = pix0 + ntl * 16 + nin;
    int b = n >> 14, rem = n & 16383;
    float* op = out + (size_t)b * 128 * 16384 + rem;
    v8f cv = (ntl == 0) ? c0 : (ntl == 1) ? c1 : (ntl == 2) ? c2 : c3;
#pragma unroll
    for (int r = 0; r < 8; ++r) {
      int oc = ocb + r;
      op[(size_t)oc * 16384] = cv[r] + conv_b[oc];
    }
  }
}

// ---------------------------------------------------------------------------
extern "C" void kernel_launch(void* const* d_in, const int* in_sizes, int n_in,
                              void* d_out, int out_size, void* d_ws, size_t ws_size,
                              hipStream_t stream) {
  const float* x      = (const float*)d_in[0];
  const float* off_w  = (const float*)d_in[1];
  const float* off_b  = (const float*)d_in[2];
  const float* conv_w = (const float*)d_in[3];
  const float* conv_b = (const float*)d_in[4];
  float* out = (float*)d_out;

  char* ws = (char*)d_ws;
  float* offs = (float*)ws;                                    // 9,437,184 B
  unsigned short* pA_conv = (unsigned short*)(ws + 9437184);   //   147,456 B
  unsigned short* pA_off  = (unsigned short*)(ws + 9437184 + 147456); // 36,864 B

  pack_weights<<<360, 256, 0, stream>>>(off_w, conv_w, pA_conv, pA_off);
  offset_conv_wmma<<<NPIX / PPB, 256, 0, stream>>>(x, pA_off, off_b, offs);
  deform_gemm_wmma<<<NPIX / PPB, 256, 0, stream>>>(x, offs, pA_conv, conv_b, out);
}
